// Attention_60584808677611
// MI455X (gfx1250) — matrix operand order repaired and checked by reading, not compiled, not run on hardware
//
#include <hip/hip_runtime.h>
#include <hip/hip_bf16.h>

// ---------------------------------------------------------------------------
// Types for CDNA5 WMMA (wave32, 16x16x32 bf16 -> f32)
// ---------------------------------------------------------------------------
typedef __bf16 bf16_t;
typedef __attribute__((ext_vector_type(16))) __bf16    v16bf;
typedef __attribute__((ext_vector_type(8)))  float     v8f;
typedef __attribute__((ext_vector_type(4)))  unsigned  v4u;   // 16B, for b128
typedef __attribute__((ext_vector_type(2)))  unsigned  v2u;   // 8B

union V16 {        // one WMMA 16-element bf16 fragment = 2 x b128
  v4u   u[2];
  v16bf v;
};
union V4B {        // 4 bf16 = 8B
  v2u    u;
  bf16_t e[4];
};
union V8B {        // 8 bf16 = 16B
  v4u    u;
  bf16_t e[8];
};

#define B_   2
#define N_   2048
#define DIM_ 1024
#define H_   16
#define DH_  64
#define M_   2048
#define KV_  4097          // 1 null + M xl + N current
#define KVP_ 4128          // padded to multiple of 32
#define ROWS_ (B_ * N_)    // 4096
#define BH_  (B_ * H_)     // 32

// Q scale folded with log2(e) so softmax runs in base-2 (v_exp_f32 native)
#define QSCALE_ 0.18033688011112042f   // (1/8) * log2(e)

__device__ __forceinline__ bf16_t f2bf(float f) {
  unsigned u = __builtin_bit_cast(unsigned, f);
  u += 0x7FFFu + ((u >> 16) & 1u);   // round-to-nearest-even
  unsigned short hs = (unsigned short)(u >> 16);
  return __builtin_bit_cast(bf16_t, hs);
}
__device__ __forceinline__ float bf2f(bf16_t b) {
  unsigned short hs = __builtin_bit_cast(unsigned short, b);
  unsigned u = ((unsigned)hs) << 16;
  return __builtin_bit_cast(float, u);
}
__device__ __forceinline__ v8f v8f_zero() {
  v8f z;
#pragma unroll
  for (int i = 0; i < 8; i++) z[i] = 0.0f;
  return z;
}
__device__ __forceinline__ float rmax16(float v) {
#pragma unroll
  for (int off = 1; off < 16; off <<= 1) v = fmaxf(v, __shfl_xor(v, off, 32));
  return v;
}
__device__ __forceinline__ float rsum16(float v) {
#pragma unroll
  for (int off = 1; off < 16; off <<= 1) v += __shfl_xor(v, off, 32);
  return v;
}

// ---------------------------------------------------------------------------
// CDNA5 async global->LDS copy (ASYNCcnt-tracked), inline asm.
// LDS address = low 32 bits of the generic shared pointer (flat LDS aperture).
// ---------------------------------------------------------------------------
__device__ __forceinline__ void async_load_b128(void* lds_ptr,
                                                const void* gptr) {
  unsigned lds_addr = (unsigned)(unsigned long long)lds_ptr;
  unsigned long long ga = (unsigned long long)gptr;
  asm volatile("global_load_async_to_lds_b128 %0, %1, off"
               :
               : "v"(lds_addr), "v"(ga)
               : "memory");
}
__device__ __forceinline__ void wait_async0() {
  asm volatile("s_wait_asynccnt 0x0" ::: "memory");
}

// ---------------------------------------------------------------------------
// 1) f32 -> bf16 conversion (weights)
// ---------------------------------------------------------------------------
__global__ void cvt_f32_bf16_kernel(const float* __restrict__ in,
                                    bf16_t* __restrict__ out, int n) {
  int i = blockIdx.x * 256 + threadIdx.x;
  if (i < n) out[i] = f2bf(in[i]);
}

// ---------------------------------------------------------------------------
// 2) RMSNorm: x[ROWS_,DIM_] -> xn bf16
// ---------------------------------------------------------------------------
__global__ __launch_bounds__(256) void rmsnorm_kernel(
    const float* __restrict__ x, const float* __restrict__ w,
    bf16_t* __restrict__ xn) {
  int row = blockIdx.x;
  int tid = threadIdx.x;
  const float* xr = x + (size_t)row * DIM_;
  __shared__ float red[256];
  float s = 0.f;
  for (int i = tid; i < DIM_; i += 256) { float v = xr[i]; s += v * v; }
  red[tid] = s;
  __syncthreads();
  for (int st = 128; st > 0; st >>= 1) {
    if (tid < st) red[tid] += red[tid + st];
    __syncthreads();
  }
  float inv = rsqrtf(red[0] * (1.0f / DIM_) + 1e-6f);
  for (int i = tid; i < DIM_; i += 256)
    xn[(size_t)row * DIM_ + i] = f2bf(xr[i] * inv * w[i]);
}

// ---------------------------------------------------------------------------
// 3) Tiled bf16 WMMA GEMM: C[M,N] = A[M,K] @ B[K,N], row-major, C f32.
//    128x128x32 tiles, 8 waves (4x2), wave = 32x64 = 2x4 wmma tiles.
//    Double-buffered LDS; A staged with GLOBAL_LOAD_ASYNC_TO_LDS_B128,
//    B staged transposed through VGPRs (hidden behind current-tile WMMAs).
// ---------------------------------------------------------------------------
__global__ __launch_bounds__(256) void gemm_bf16_kernel(
    const bf16_t* __restrict__ A, const bf16_t* __restrict__ Bm,
    float* __restrict__ C, int Mdim, int Ndim, int Kdim) {
  __shared__ bf16_t As[2][128][40];    // row stride 80B (16B multiple)
  __shared__ bf16_t BsT[2][128][40];   // [col][k]
  const int nTilesN = Ndim >> 7;
  const int bm = blockIdx.x / nTilesN;
  const int bn = blockIdx.x % nTilesN;
  const int tid = threadIdx.x;
  const int lane = tid & 31;
  const int wave = tid >> 5;
  const int wm = wave >> 1;        // 0..3 -> row offset wm*32
  const int wn = wave & 1;         // 0..1 -> col offset wn*64
  const int half = lane >> 4;
  const int l16 = lane & 15;
  const int rowBase = bm << 7;
  const int colBase = bn << 7;
  const int arow = tid >> 1, acol = (tid & 1) << 4;   // A: 16 bf16 per thread
  const int brow = tid >> 3, bcol = (tid & 7) << 4;   // B: 16 bf16 per thread

  v8f acc[2][4];
#pragma unroll
  for (int i = 0; i < 2; i++)
#pragma unroll
    for (int j = 0; j < 4; j++) acc[i][j] = v8f_zero();

  auto stageA = [&](int buf, int k0) {
    const bf16_t* asrc = A + (size_t)(rowBase + arow) * Kdim + k0 + acol;
    async_load_b128(&As[buf][arow][acol], asrc);
    async_load_b128(&As[buf][arow][acol + 8], asrc + 8);
  };
  auto stageB = [&](int buf, int k0) {
    const bf16_t* bsrc = Bm + (size_t)(k0 + brow) * Ndim + colBase + bcol;
    V16 bt;
    bt.u[0] = *(const v4u*)bsrc;
    bt.u[1] = *(const v4u*)(bsrc + 8);
#pragma unroll
    for (int i = 0; i < 16; i++) BsT[buf][bcol + i][brow] = bt.v[i];
  };

  const int nk = Kdim >> 5;
  stageA(0, 0);
  stageB(0, 0);

  for (int it = 0; it < nk; ++it) {
    const int buf = it & 1;
    wait_async0();        // this wave's async A-tile writes done
    __syncthreads();      // all waves' staging (A async + B ds) visible
    if (it + 1 < nk) {    // overlap next-tile staging with compute
      stageA(buf ^ 1, (it + 1) << 5);
      stageB(buf ^ 1, (it + 1) << 5);
    }

    // ---- fragments: 2 x ds_load_b128 each ----
    V16 af[2];
#pragma unroll
    for (int tm = 0; tm < 2; tm++) {
      int row = (wm << 5) + (tm << 4) + l16;
      af[tm].u[0] = *(const v4u*)&As[buf][row][(half << 3)];
      af[tm].u[1] = *(const v4u*)&As[buf][row][16 + (half << 3)];
    }
    V16 bf[4];
#pragma unroll
    for (int tn = 0; tn < 4; tn++) {
      int col = (wn << 6) + (tn << 4) + l16;
      bf[tn].u[0] = *(const v4u*)&BsT[buf][col][(half << 3)];
      bf[tn].u[1] = *(const v4u*)&BsT[buf][col][16 + (half << 3)];
    }
#pragma unroll
    for (int tm = 0; tm < 2; tm++)
#pragma unroll
      for (int tn = 0; tn < 4; tn++)
        acc[tm][tn] = __builtin_amdgcn_wmma_f32_16x16x32_bf16(
            false, af[tm].v, false, bf[tn].v, (short)0, acc[tm][tn],
            false, false);
    __syncthreads();      // readers done before buf is restaged next+1 iter
  }

#pragma unroll
  for (int tm = 0; tm < 2; tm++)
#pragma unroll
    for (int tn = 0; tn < 4; tn++)
#pragma unroll
      for (int r = 0; r < 8; r++) {
        int row = rowBase + (wm << 5) + (tm << 4) + r + (half << 3);
        int col = colBase + (wn << 6) + (tn << 4) + l16;
        C[(size_t)row * Ndim + col] = acc[tm][tn][r];
      }
}

// ---------------------------------------------------------------------------
// 4) Mix logits: mixlog[row][h] = dot(xn[row,:], Wmix[:,h])  (N=16 -> VALU)
// ---------------------------------------------------------------------------
__global__ __launch_bounds__(256) void mix_logits_kernel(
    const bf16_t* __restrict__ xn, const float* __restrict__ Wmix,
    float* __restrict__ mixlog) {
  int idx = blockIdx.x * 256 + threadIdx.x;  // ROWS_*H_ = 65536
  if (idx >= ROWS_ * H_) return;
  int h = idx & 15;
  int row = idx >> 4;
  const bf16_t* xr = xn + (size_t)row * DIM_;
  float s = 0.f;
  for (int i = 0; i < DIM_; i += 8) {
    V8B xv;
    xv.u = *(const v4u*)(xr + i);
#pragma unroll
    for (int j = 0; j < 8; j++)
      s += bf2f(xv.e[j]) * Wmix[(size_t)(i + j) * H_ + h];
  }
  mixlog[idx] = s;
}

// ---------------------------------------------------------------------------
// 5) RoPE on Q (folds softmax scale * log2(e)): qproj f32 -> qr bf16 [b,h,n,64]
// ---------------------------------------------------------------------------
__global__ __launch_bounds__(256) void rope_q_kernel(
    const float* __restrict__ qproj, const float* __restrict__ rq,
    bf16_t* __restrict__ qr) {
  int idx = blockIdx.x * 256 + threadIdx.x;  // B*H*N*32
  if (idx >= B_ * H_ * N_ * 32) return;
  int d = idx & 31;
  int n = (idx >> 5) & (N_ - 1);
  int h = (idx >> 16) & (H_ - 1);
  int b = idx >> 20;
  size_t src = (size_t)(b * N_ + n) * (H_ * DH_) + h * DH_ + d;
  float q1 = qproj[src];
  float q2 = qproj[src + 32];
  float p1 = rq[n * DH_ + d], p2 = rq[n * DH_ + d + 32];
  float o1 = q1 * __cosf(p1) - q2 * __sinf(p1);
  float o2 = q2 * __cosf(p2) + q1 * __sinf(p2);
  size_t dst = (((size_t)b * H_ + h) * N_ + n) * DH_ + d;
  qr[dst] = f2bf(o1 * QSCALE_);
  qr[dst + 32] = f2bf(o2 * QSCALE_);
}

// ---------------------------------------------------------------------------
// 6) KV prep: build padded roped-K (row-major) and mixed-V bf16 caches,
//    write next_xl_memories (f32) and orig_v (f32) outputs.
//    Concat order: [xl (0..2047)] [null (2048)] [current (2049..4096)] [pad]
// ---------------------------------------------------------------------------
__global__ __launch_bounds__(256) void kv_prep_kernel(
    const float* __restrict__ xl,       // [2,B,H,M,DH]
    const float* __restrict__ nullkv,   // [2,H,DH]
    const float* __restrict__ kvproj,   // [B,N, 2*H*DH]
    const float* __restrict__ mixlog,   // [B,N,H]
    const float* __restrict__ vres,     // [B,H,N,DH]
    const float* __restrict__ rk,       // [KV_,DH]
    bf16_t* __restrict__ kcat, bf16_t* __restrict__ vcat,
    float* __restrict__ nxk, float* __restrict__ nxv,  // [B,H,1+N,DH]
    float* __restrict__ origv) {                        // [B,H,N,DH]
  int idx = blockIdx.x * 256 + threadIdx.x;  // B*H*KVP_*32
  if (idx >= B_ * H_ * KVP_ * 32) return;
  int d = idx & 31;
  int t = idx >> 5;
  int pos = t % KVP_;
  t /= KVP_;
  int h = t & (H_ - 1);
  int b = t >> 4;
  size_t cdst = (((size_t)b * H_ + h) * KVP_ + pos) * DH_ + d;

  if (pos > 4096) {  // padding
    kcat[cdst] = f2bf(0.f); kcat[cdst + 32] = f2bf(0.f);
    vcat[cdst] = f2bf(0.f); vcat[cdst + 32] = f2bf(0.f);
    return;
  }

  float k1, k2, v1, v2;
  if (pos < M_) {  // xl memories
    size_t base = (((size_t)b * H_ + h) * M_ + pos) * DH_ + d;
    const float* xlv = xl + (size_t)B_ * H_ * M_ * DH_;
    k1 = xl[base];  k2 = xl[base + 32];
    v1 = xlv[base]; v2 = xlv[base + 32];
  } else if (pos == M_) {  // null kv -> next_xl slot 0
    size_t nb = (size_t)h * DH_ + d;
    k1 = nullkv[nb];                     k2 = nullkv[nb + 32];
    v1 = nullkv[(size_t)H_ * DH_ + nb];  v2 = nullkv[(size_t)H_ * DH_ + nb + 32];
    size_t xd = (((size_t)b * H_ + h) * (1 + N_) + 0) * DH_ + d;
    nxk[xd] = k1; nxk[xd + 32] = k2;
    nxv[xd] = v1; nxv[xd + 32] = v2;
  } else {  // current tokens
    int n = pos - (M_ + 1);
    size_t prow = (size_t)(b * N_ + n) * (2 * H_ * DH_);
    k1 = kvproj[prow + h * DH_ + d];
    k2 = kvproj[prow + h * DH_ + d + 32];
    float vo1 = kvproj[prow + H_ * DH_ + h * DH_ + d];
    float vo2 = kvproj[prow + H_ * DH_ + h * DH_ + d + 32];
    float ml = mixlog[(size_t)(b * N_ + n) * H_ + h];
    float mix = 1.0f / (1.0f + __expf(-ml));
    size_t rb = (((size_t)b * H_ + h) * N_ + n) * DH_ + d;
    v1 = vo1 + mix * (vres[rb] - vo1);
    v2 = vo2 + mix * (vres[rb + 32] - vo2);
    origv[rb] = vo1; origv[rb + 32] = vo2;
    size_t xd = (((size_t)b * H_ + h) * (1 + N_) + 1 + n) * DH_ + d;
    nxk[xd] = k1; nxk[xd + 32] = k2;
    nxv[xd] = v1; nxv[xd + 32] = v2;
  }
  // RoPE on k (next_xl stores pre-rope k, as in the reference)
  float p1 = rk[pos * DH_ + d], p2 = rk[pos * DH_ + d + 32];
  float rk1 = k1 * __cosf(p1) - k2 * __sinf(p1);
  float rk2 = k2 * __cosf(p2) + k1 * __sinf(p2);
  kcat[cdst] = f2bf(rk1); kcat[cdst + 32] = f2bf(rk2);
  vcat[cdst] = f2bf(v1);  vcat[cdst + 32] = f2bf(v2);
}

// ---------------------------------------------------------------------------
// 6b) Transpose V cache: vcat [BH, KVP_, 64] -> vcatT [BH, 64, KVP_]
//     so attention PV B-fragments become contiguous b128 loads.
// ---------------------------------------------------------------------------
__global__ __launch_bounds__(256) void transpose_v_kernel(
    const bf16_t* __restrict__ in, bf16_t* __restrict__ out) {
  __shared__ bf16_t tile[32][40];   // 32 keys x 32 d, 80B stride
  int blk = blockIdx.x;             // BH * 129 * 2
  int dblk = blk & 1;
  int kblk = (blk >> 1) % (KVP_ / 32);
  int bh = blk / (2 * (KVP_ / 32));
  int tid = threadIdx.x;
  int r = tid >> 3;                 // 0..31
  int c = (tid & 7) << 2;           // 0,4,..,28
  const bf16_t* src =
      in + (((size_t)bh * KVP_ + kblk * 32 + r) * DH_) + dblk * 32 + c;
  *(v2u*)&tile[r][c] = *(const v2u*)src;   // 8B load/store
  __syncthreads();
  // write: thread handles d-row r, keys c..c+3
  V4B o;
#pragma unroll
  for (int i = 0; i < 4; i++) o.e[i] = tile[c + i][r];
  bf16_t* dst =
      out + (((size_t)bh * DH_ + dblk * 32 + r) * KVP_) + kblk * 32 + c;
  *(v2u*)dst = o.u;
}

// ---------------------------------------------------------------------------
// 7) Flash attention: one wave per (b,h,16-row q tile). Online softmax in
//    base-2, WMMA for QK^T and PV; all fragments built from b128 loads.
// ---------------------------------------------------------------------------
__global__ __launch_bounds__(256) void flash_attn_kernel(
    const bf16_t* __restrict__ Q,    // [b,h,N,64] roped, * (scale*log2e)
    const bf16_t* __restrict__ Kc,   // [b,h,KVP_,64] roped
    const bf16_t* __restrict__ VcT,  // [b,h,64,KVP_]
    bf16_t* __restrict__ Obf) {      // [ROWS_, H_*DH_]
  __shared__ bf16_t Plds[8][16][40];  // per-wave P tile, 80B row stride
  const int tid = threadIdx.x;
  const int lane = tid & 31;
  const int wave = tid >> 5;
  const int gw = blockIdx.x * 8 + wave;   // 0..4095
  const int qt = gw & 127;
  const int h = (gw >> 7) & (H_ - 1);
  const int b = gw >> 11;
  const int half = lane >> 4;
  const int l16 = lane & 15;

  const bf16_t* q  = Q + (((size_t)b * H_ + h) * N_ + qt * 16) * DH_;
  const bf16_t* kc = Kc + ((size_t)b * H_ + h) * KVP_ * DH_;
  const bf16_t* vt = VcT + ((size_t)b * H_ + h) * DH_ * KVP_;

  // Q A-fragments: two b128 loads per 32-wide d-chunk
  V16 aq[2];
#pragma unroll
  for (int c = 0; c < 2; c++) {
    const bf16_t* qp = q + (size_t)l16 * DH_ + c * 32 + (half << 3);
    aq[c].u[0] = *(const v4u*)qp;
    aq[c].u[1] = *(const v4u*)(qp + 16);
  }

  v8f o[4];
#pragma unroll
  for (int t = 0; t < 4; t++) o[t] = v8f_zero();
  float mrow[8], lrow[8];
#pragma unroll
  for (int r = 0; r < 8; r++) { mrow[r] = -1e30f; lrow[r] = 0.f; }

  for (int kb = 0; kb < KVP_; kb += 32) {
    if (kb + 32 < KVP_) {
      __builtin_prefetch(kc + (size_t)(kb + 32 + l16) * DH_, 0, 3);
      __builtin_prefetch(vt + (size_t)l16 * KVP_ + kb + 32, 0, 3);
    }
    // ---- S = Q K^T for two 16-key subtiles ----
    v8f s[2];
#pragma unroll
    for (int sub = 0; sub < 2; sub++) {
      v8f ss = v8f_zero();
      int key = kb + sub * 16 + l16;
#pragma unroll
      for (int c = 0; c < 2; c++) {
        const bf16_t* krow = kc + (size_t)key * DH_ + c * 32 + (half << 3);
        V16 bk;
        bk.u[0] = *(const v4u*)krow;
        bk.u[1] = *(const v4u*)(krow + 16);
        ss = __builtin_amdgcn_wmma_f32_16x16x32_bf16(
            false, aq[c].v, false, bk.v, (short)0, ss, false, false);
      }
      s[sub] = ss;
    }
    // mask padding columns (key >= KV_)
    if (kb + l16 >= KV_) {
#pragma unroll
      for (int r = 0; r < 8; r++) s[0][r] = -1e30f;
    }
    if (kb + 16 + l16 >= KV_) {
#pragma unroll
      for (int r = 0; r < 8; r++) s[1][r] = -1e30f;
    }
    // ---- online softmax per row (base-2) ----
    float corr[8];
#pragma unroll
    for (int r = 0; r < 8; r++) {
      float tmax = rmax16(fmaxf(s[0][r], s[1][r]));
      float nm = fmaxf(mrow[r], tmax);
      corr[r] = exp2f(mrow[r] - nm);
      mrow[r] = nm;
      float p0 = exp2f(s[0][r] - nm);
      float p1 = exp2f(s[1][r] - nm);
      lrow[r] = lrow[r] * corr[r] + rsum16(p0 + p1);
      Plds[wave][r + (half << 3)][l16] = f2bf(p0);
      Plds[wave][r + (half << 3)][16 + l16] = f2bf(p1);
    }
#pragma unroll
    for (int t = 0; t < 4; t++)
#pragma unroll
      for (int r = 0; r < 8; r++) o[t][r] *= corr[r];
    // ---- P (A-layout) from LDS: 2 x ds_load_b128 ----
    V16 pa;
    {
      const bf16_t* pr = &Plds[wave][l16][(half << 3)];
      pa.u[0] = *(const v4u*)pr;
      pa.u[1] = *(const v4u*)(pr + 16);
    }
    // ---- o += P @ V  (V fragments contiguous from vcatT) ----
#pragma unroll
    for (int t = 0; t < 4; t++) {
      const bf16_t* vrow =
          vt + (size_t)(t * 16 + l16) * KVP_ + kb + (half << 3);
      V16 bv;
      bv.u[0] = *(const v4u*)vrow;
      bv.u[1] = *(const v4u*)(vrow + 16);
      o[t] = __builtin_amdgcn_wmma_f32_16x16x32_bf16(
          false, pa.v, false, bv.v, (short)0, o[t], false, false);
    }
  }
  // ---- epilogue: normalize (one rcp per row) and store bf16 ----
#pragma unroll
  for (int r = 0; r < 8; r++) {
    float inv = 1.0f / lrow[r];
    int n = qt * 16 + r + (half << 3);
#pragma unroll
    for (int t = 0; t < 4; t++) {
      int col = h * DH_ + t * 16 + l16;
      Obf[(size_t)(b * N_ + n) * (H_ * DH_) + col] = f2bf(o[t][r] * inv);
    }
  }
}

// ---------------------------------------------------------------------------
// Host launcher
// ---------------------------------------------------------------------------
extern "C" void kernel_launch(void* const* d_in, const int* in_sizes, int n_in,
                              void* d_out, int out_size, void* d_ws,
                              size_t ws_size, hipStream_t stream) {
  const float* x      = (const float*)d_in[0];
  const float* vres   = (const float*)d_in[1];
  const float* xl     = (const float*)d_in[2];
  const float* rq     = (const float*)d_in[3];
  const float* rk     = (const float*)d_in[4];
  const float* norm_w = (const float*)d_in[5];
  const float* Wq     = (const float*)d_in[6];
  const float* Wkv    = (const float*)d_in[7];
  const float* Wout   = (const float*)d_in[8];
  const float* nullkv = (const float*)d_in[9];
  const float* Wmix   = (const float*)d_in[10];

  float* out   = (float*)d_out;                               // [B,N,DIM]
  float* nxk   = out + (size_t)ROWS_ * DIM_;                  // [B,H,1+N,DH]
  float* nxv   = nxk + (size_t)B_ * H_ * (1 + N_) * DH_;
  float* origv = nxv + (size_t)B_ * H_ * (1 + N_) * DH_;      // [B,H,N,DH]

  // workspace carve-out
  size_t off = 0;
  char* ws = (char*)d_ws;
  auto alloc = [&](size_t bytes) -> void* {
    void* p = ws + off;
    off += (bytes + 255) & ~(size_t)255;
    return p;
  };
  bf16_t* xn_bf   = (bf16_t*)alloc((size_t)ROWS_ * DIM_ * 2);
  bf16_t* Wq_bf   = (bf16_t*)alloc((size_t)DIM_ * (H_ * DH_) * 2);
  bf16_t* Wkv_bf  = (bf16_t*)alloc((size_t)DIM_ * (2 * H_ * DH_) * 2);
  bf16_t* Wout_bf = (bf16_t*)alloc((size_t)(H_ * DH_) * DIM_ * 2);
  float*  qproj   = (float*)alloc((size_t)ROWS_ * (H_ * DH_) * 4);
  float*  kvproj  = (float*)alloc((size_t)ROWS_ * (2 * H_ * DH_) * 4);
  float*  mixlog  = (float*)alloc((size_t)ROWS_ * H_ * 4);
  bf16_t* qr      = (bf16_t*)alloc((size_t)BH_ * N_ * DH_ * 2);
  bf16_t* kcat    = (bf16_t*)alloc((size_t)BH_ * KVP_ * DH_ * 2);
  bf16_t* vcat    = (bf16_t*)alloc((size_t)BH_ * KVP_ * DH_ * 2);
  bf16_t* vcatT   = (bf16_t*)alloc((size_t)BH_ * DH_ * KVP_ * 2);
  bf16_t* obf     = (bf16_t*)alloc((size_t)ROWS_ * (H_ * DH_) * 2);

  // 1) weight conversion
  {
    int n1 = DIM_ * H_ * DH_;       // 1M
    int n2 = DIM_ * 2 * H_ * DH_;   // 2M
    cvt_f32_bf16_kernel<<<(n1 + 255) / 256, 256, 0, stream>>>(Wq, Wq_bf, n1);
    cvt_f32_bf16_kernel<<<(n2 + 255) / 256, 256, 0, stream>>>(Wkv, Wkv_bf, n2);
    cvt_f32_bf16_kernel<<<(n1 + 255) / 256, 256, 0, stream>>>(Wout, Wout_bf, n1);
  }
  // 2) RMSNorm
  rmsnorm_kernel<<<ROWS_, 256, 0, stream>>>(x, norm_w, xn_bf);
  // 3) Q and KV projections (WMMA GEMMs)
  gemm_bf16_kernel<<<(ROWS_ / 128) * ((H_ * DH_) / 128), 256, 0, stream>>>(
      xn_bf, Wq_bf, qproj, ROWS_, H_ * DH_, DIM_);
  gemm_bf16_kernel<<<(ROWS_ / 128) * ((2 * H_ * DH_) / 128), 256, 0, stream>>>(
      xn_bf, Wkv_bf, kvproj, ROWS_, 2 * H_ * DH_, DIM_);
  // 4) mix logits
  mix_logits_kernel<<<(ROWS_ * H_) / 256, 256, 0, stream>>>(xn_bf, Wmix, mixlog);
  // 5) RoPE on Q
  rope_q_kernel<<<(B_ * H_ * N_ * 32) / 256, 256, 0, stream>>>(qproj, rq, qr);
  // 6) KV prep (rope K, value lerp, concat, side outputs)
  {
    int total = B_ * H_ * KVP_ * 32;
    kv_prep_kernel<<<(total + 255) / 256, 256, 0, stream>>>(
        xl, nullkv, kvproj, mixlog, vres, rk, kcat, vcat, nxk, nxv, origv);
  }
  // 6b) transpose V cache for contiguous PV fragments
  transpose_v_kernel<<<BH_ * (KVP_ / 32) * 2, 256, 0, stream>>>(vcat, vcatT);
  // 7) attention
  flash_attn_kernel<<<(BH_ * (N_ / 16)) / 8, 256, 0, stream>>>(
      qr, kcat, vcatT, obf);
  // 8) output projection -> d_out
  gemm_bf16_kernel<<<(ROWS_ / 128) * (DIM_ / 128), 256, 0, stream>>>(
      obf, Wout_bf, out, ROWS_, DIM_, DIM_);
}
